// Transformer_8959301779846
// MI455X (gfx1250) — matrix operand order repaired and checked by reading, not compiled, not run on hardware
//
#include <hip/hip_runtime.h>
#include <hip/hip_bf16.h>
#include <math.h>

#define N_TOTAL   131072
#define NODE_F    64
#define NORM_DIM  32
#define DIM       512
#define NUM_CLS   8
#define B_GRAPHS  16

typedef __attribute__((ext_vector_type(16))) _Float16 v16h;
typedef __attribute__((ext_vector_type(8)))  _Float16 v8h;
typedef __attribute__((ext_vector_type(8)))  float    v8f;

union AFrag { v16h v; v8h h2[2]; };

__device__ __forceinline__ v8f wmma_f16(v16h a, v16h b, v8f c) {
    // D = A(16x32 f16) * B(32x16 f16) + C(16x16 f32)
    return __builtin_amdgcn_wmma_f32_16x16x32_f16(
        /*neg_a=*/false, a, /*neg_b=*/false, b,
        /*c_mod=*/(short)0, c, /*reuse_a=*/false, /*reuse_b=*/false);
}

// Build an A-fragment from f16 row-major LDS tile.
// A layout (16x32 f16): lane L -> m=L%16, kh=L/16; elem e<8 -> k=k0+kh*8+e,
// elem e>=8 -> k=k0+16+kh*8+(e-8).  => two contiguous 8-half (16B) loads.
__device__ __forceinline__ v16h load_afrag(const _Float16* base, int m,
                                           int stride, int k0, int kh) {
    const _Float16* p = base + m * stride + k0 + kh * 8;
    AFrag u;
    u.h2[0] = *reinterpret_cast<const v8h*>(p);
    u.h2[1] = *reinterpret_cast<const v8h*>(p + 16);
    return u.v;
}

// ---------------------------------------------------------------------------
// Kernel 0: pack fp32 weight matrix (K x N) into f16 WMMA B-fragment layout.
// B layout (32x16 f16 per 16-col tile, per 32-k step):
//   lane L -> n = tile*16 + L%16,  elem e<8 -> k = ks*32 + (L/16)*8 + e, e>=8 -> k = ks*32 + 16 + (L/16)*8 + (e-8)
// Linear index: ((tile*KS + ks)*32 + L)*16 + e
// ---------------------------------------------------------------------------
__global__ void pack_kernel(const float* __restrict__ W, _Float16* __restrict__ dst,
                            int K, int N, int Ntiles) {
    int KS = K >> 5;
    int total = Ntiles * KS * 512;
    for (int idx = blockIdx.x * blockDim.x + threadIdx.x; idx < total;
         idx += gridDim.x * blockDim.x) {
        int e    = idx & 15;
        int L    = (idx >> 4) & 31;
        int ks   = (idx >> 9) % KS;
        int tile = idx / (512 * KS);
        int n = tile * 16 + (L & 15);
        int k = ks * 32 + (e < 8 ? (L >> 4) * 8 + e : 16 + (L >> 4) * 8 + (e - 8));
        float v = (n < N) ? W[(size_t)k * N + n] : 0.0f;
        dst[idx] = (_Float16)v;
    }
}

// ---------------------------------------------------------------------------
// Kernel 1a/1b: deterministic BatchNorm statistics over x[:, :32]
// ---------------------------------------------------------------------------
#define BN_BLOCKS 256
__global__ void bn_partial(const float* __restrict__ x, float* __restrict__ part) {
    __shared__ float ssum[256], ssq[256];
    int t = threadIdx.x;
    int c = t & 31;          // column
    int g = t >> 5;          // row-group 0..7
    const int rows = N_TOTAL / BN_BLOCKS;   // 512
    int r0 = blockIdx.x * rows;
    float s = 0.f, q = 0.f;
    for (int r = g; r < rows; r += 8) {
        float v = x[(size_t)(r0 + r) * NODE_F + c];
        s += v; q += v * v;
    }
    ssum[t] = s; ssq[t] = q;
    __syncthreads();
    if (t < 32) {
        float S = 0.f, Q = 0.f;
        for (int gg = 0; gg < 8; ++gg) { S += ssum[gg * 32 + t]; Q += ssq[gg * 32 + t]; }
        part[blockIdx.x * 64 + t]      = S;
        part[blockIdx.x * 64 + 32 + t] = Q;
    }
}

__global__ void bn_final(const float* __restrict__ part,
                         const float* __restrict__ gamma,
                         const float* __restrict__ beta,
                         float* __restrict__ scaleshift) {
    int c = threadIdx.x;
    if (c >= 32) return;
    float S = 0.f, Q = 0.f;
    for (int b = 0; b < BN_BLOCKS; ++b) { S += part[b * 64 + c]; Q += part[b * 64 + 32 + c]; }
    float mean = S / (float)N_TOTAL;
    float var  = Q / (float)N_TOTAL - mean * mean;
    float inv  = rsqrtf(var + 1e-5f);
    float sc   = gamma[c] * inv;
    scaleshift[c]      = sc;
    scaleshift[32 + c] = beta[c] - mean * sc;
}

// ---------------------------------------------------------------------------
// Kernel 2: fused per-node pipeline.  One wave32 = one 16-row tile.
// norm -> efd GEMM + gelu(geo1) GEMM -> geo2 GEMM -> rms_norm ->
// cls GEMM (nuclei logits) + attn1 GEMM -> tanh -> dot(attn_w2) -> raw score
// ---------------------------------------------------------------------------
#define WAVES 2
#define TPB (WAVES * 32)

__global__ void __launch_bounds__(TPB) fused_kernel(
    const float* __restrict__ x,
    const float* __restrict__ scaleshift,
    const _Float16* __restrict__ pk_efd,
    const _Float16* __restrict__ pk_geo1,
    const _Float16* __restrict__ pk_geo2,
    const _Float16* __restrict__ pk_attn1,
    const _Float16* __restrict__ pk_cls,
    const float* __restrict__ efd_b,
    const float* __restrict__ geo_b1,
    const float* __restrict__ geo_b2,
    const float* __restrict__ fnw,
    const float* __restrict__ cls_b,
    const float* __restrict__ attn_b1,
    const float* __restrict__ attn_w2,
    const float* __restrict__ attn_b2,
    float* __restrict__ nuclei,
    float* __restrict__ scores) {

    __shared__ _Float16 lds_x[WAVES][16][NODE_F];   // normalized input tile
    __shared__ _Float16 lds_g[WAVES][16][DIM / 2];  // gelu(geo1) tile
    __shared__ _Float16 lds_h[WAVES][16][DIM];      // hidden tile
    __shared__ float    rowsum[WAVES][16];

    const int tid  = threadIdx.x;
    const int w    = tid >> 5;
    const int lane = tid & 31;
    const int n16  = lane & 15;
    const int kh   = lane >> 4;
    const int T    = blockIdx.x * WAVES + w;
    const int row0 = T * 16;

    const v16h* bf_efd   = (const v16h*)pk_efd;
    const v16h* bf_geo1  = (const v16h*)pk_geo1;
    const v16h* bf_geo2  = (const v16h*)pk_geo2;
    const v16h* bf_attn1 = (const v16h*)pk_attn1;
    const v16h* bf_cls   = (const v16h*)pk_cls;

    // ---- stage normalized x tile into LDS (f16) ----
    for (int idx = lane; idx < 16 * NODE_F; idx += 32) {
        int r = idx >> 6, c = idx & 63;
        float v = x[(size_t)(row0 + r) * NODE_F + c];
        if (c < NORM_DIM) v = v * scaleshift[c] + scaleshift[32 + c];
        lds_x[w][r][c] = (_Float16)v;
    }
    __syncthreads();

    // ---- geo1: 16x256 = x[:,32:64] @ geo_w1, gelu, to LDS ----
    {
        const v16h a_geo = load_afrag(&lds_x[w][0][0], n16, NODE_F, 32, kh);
        for (int t = 0; t < 16; ++t) {
            v8f acc = {};
            v16h b = bf_geo1[t * 32 + lane];
            acc = wmma_f16(a_geo, b, acc);
            int n = t * 16 + n16;
            float b1 = geo_b1[n];
            for (int r = 0; r < 8; ++r) {
                float z = acc[r] + b1;
                float g = 0.5f * z * (1.0f + erff(z * 0.70710678118654752f));
                lds_g[w][r + 8 * kh][n] = (_Float16)g;
            }
        }
    }
    __syncthreads();

    // ---- h = norm @ efd_w + gelu @ geo_w2 + biases; track row sum(h^2) ----
    // Hoist all A-fragments into registers: efd A (8 VGPR) + 8 geo2 A (64 VGPR).
    float rp[8] = {};
    {
        const v16h a_efd = load_afrag(&lds_x[w][0][0], n16, NODE_F, 0, kh);
        v16h a_g[8];
#pragma unroll
        for (int ks = 0; ks < 8; ++ks)
            a_g[ks] = load_afrag(&lds_g[w][0][0], n16, DIM / 2, ks * 32, kh);

        for (int t = 0; t < 32; ++t) {
            v8f acc = {};
            acc = wmma_f16(a_efd, bf_efd[t * 32 + lane], acc);
#pragma unroll
            for (int ks = 0; ks < 8; ++ks)
                acc = wmma_f16(a_g[ks], bf_geo2[(t * 8 + ks) * 32 + lane], acc);

            int n = t * 16 + n16;
            float bb = efd_b[n] + geo_b2[n];
            for (int r = 0; r < 8; ++r) {
                float h = acc[r] + bb;
                rp[r] += h * h;
                lds_h[w][r + 8 * kh][n] = (_Float16)h;
            }
        }
    }
    // deterministic cross-lane reduction within each 16-lane half
    for (int r = 0; r < 8; ++r) {
        float v = rp[r];
        for (int off = 1; off < 16; off <<= 1) v += __shfl_xor(v, off, 32);
        rp[r] = v;
    }
    if (n16 == 0)
        for (int r = 0; r < 8; ++r) rowsum[w][kh * 8 + r] = rp[r];
    __syncthreads();

    // ---- rms_norm fold: h *= rsqrt(mean(h^2)+eps) * final_norm_w ----
    if (tid < WAVES * 16) {   // one inv-rms per row, computed once
        int ww = tid >> 4, rr = tid & 15;
        rowsum[ww][rr] = rsqrtf(rowsum[ww][rr] * (1.0f / (float)DIM) +
                                1.1920928955078125e-07f);
    }
    __syncthreads();
    for (int idx = lane; idx < 16 * DIM; idx += 32) {
        int r = idx >> 9, c = idx & 511;
        float hv = (float)lds_h[w][r][c] * rowsum[w][r] * fnw[c];
        lds_h[w][r][c] = (_Float16)hv;
    }
    __syncthreads();

    // ---- hoist all 16 h A-fragments (K=512) into registers (128 VGPR) ----
    v16h a_h[16];
#pragma unroll
    for (int ks = 0; ks < 16; ++ks)
        a_h[ks] = load_afrag(&lds_h[w][0][0], n16, DIM, ks * 32, kh);

    // ---- nuclei_logits = h @ cls_w + cls_b  (N padded 8 -> 16) ----
    {
        v8f acc = {};
#pragma unroll
        for (int ks = 0; ks < 16; ++ks)
            acc = wmma_f16(a_h[ks], bf_cls[ks * 32 + lane], acc);
        if (n16 < NUM_CLS) {
            float cb = cls_b[n16];
            for (int r = 0; r < 8; ++r)
                nuclei[(size_t)(row0 + r + 8 * kh) * NUM_CLS + n16] = acc[r] + cb;
        }
    }

    // ---- attn score: tanh(h @ attn_w1 + b1) . attn_w2 + b2 ----
    float sp[8] = {};
    for (int t = 0; t < 16; ++t) {
        v8f acc = {};
#pragma unroll
        for (int ks = 0; ks < 16; ++ks)
            acc = wmma_f16(a_h[ks], bf_attn1[(t * 16 + ks) * 32 + lane], acc);
        int n = t * 16 + n16;
        float b1 = attn_b1[n], w2 = attn_w2[n];
        for (int r = 0; r < 8; ++r) sp[r] += tanhf(acc[r] + b1) * w2;
    }
    for (int r = 0; r < 8; ++r) {
        float v = sp[r];
        for (int off = 1; off < 16; off <<= 1) v += __shfl_xor(v, off, 32);
        sp[r] = v;
    }
    if (n16 == 0) {
        float b2 = attn_b2[0];
        for (int r = 0; r < 8; ++r) scores[row0 + kh * 8 + r] = sp[r] + b2;
    }
}

// ---------------------------------------------------------------------------
// Kernel 3: per-segment softmax + weighted class sum (graph logits).
// ---------------------------------------------------------------------------
#define K3T 256
__global__ void seg_kernel(const int* __restrict__ seq_lens,
                           const float* __restrict__ nuclei,
                           float* __restrict__ attn,   // scores in, weights out
                           float* __restrict__ graph) {
    __shared__ float red[K3T];
    int b = blockIdx.x, t = threadIdx.x;
    int start = 0;
    for (int i = 0; i < b; ++i) start += seq_lens[i];
    int len = seq_lens[b];

    float m = -INFINITY;
    for (int i = t; i < len; i += K3T) m = fmaxf(m, attn[start + i]);
    red[t] = m; __syncthreads();
    for (int s = K3T / 2; s > 0; s >>= 1) {
        if (t < s) red[t] = fmaxf(red[t], red[t + s]);
        __syncthreads();
    }
    m = red[0]; __syncthreads();

    float sm = 0.f;
    for (int i = t; i < len; i += K3T) sm += expf(attn[start + i] - m);
    red[t] = sm; __syncthreads();
    for (int s = K3T / 2; s > 0; s >>= 1) {
        if (t < s) red[t] += red[t + s];
        __syncthreads();
    }
    float inv = 1.0f / red[0]; __syncthreads();

    float gl[NUM_CLS] = {};
    for (int i = t; i < len; i += K3T) {
        float wgt = expf(attn[start + i] - m) * inv;
        attn[start + i] = wgt;
        const float* lp = nuclei + (size_t)(start + i) * NUM_CLS;
        for (int c = 0; c < NUM_CLS; ++c) gl[c] += wgt * lp[c];
    }
    for (int c = 0; c < NUM_CLS; ++c) {
        red[t] = gl[c]; __syncthreads();
        for (int s = K3T / 2; s > 0; s >>= 1) {
            if (t < s) red[t] += red[t + s];
            __syncthreads();
        }
        if (t == 0) graph[b * NUM_CLS + c] = red[0];
        __syncthreads();
    }
}

// ---------------------------------------------------------------------------
extern "C" void kernel_launch(void* const* d_in, const int* in_sizes, int n_in,
                              void* d_out, int out_size, void* d_ws, size_t ws_size,
                              hipStream_t stream) {
    const float* x         = (const float*)d_in[0];
    const int*   seq_lens  = (const int*)d_in[1];
    const float* bn_gamma  = (const float*)d_in[2];
    const float* bn_beta   = (const float*)d_in[3];
    const float* efd_w     = (const float*)d_in[4];
    const float* efd_b     = (const float*)d_in[5];
    const float* geo_w1    = (const float*)d_in[6];
    const float* geo_b1    = (const float*)d_in[7];
    const float* geo_w2    = (const float*)d_in[8];
    const float* geo_b2    = (const float*)d_in[9];
    const float* fnw       = (const float*)d_in[10];
    const float* cls_w     = (const float*)d_in[11];
    const float* cls_b     = (const float*)d_in[12];
    const float* attn_w1   = (const float*)d_in[13];
    const float* attn_b1   = (const float*)d_in[14];
    const float* attn_w2   = (const float*)d_in[15];
    const float* attn_b2   = (const float*)d_in[16];

    char* ws = (char*)d_ws;
    float*    scaleshift = (float*)(ws + 0);          // 64 f32
    float*    bnpart     = (float*)(ws + 1024);       // 256*64 f32
    _Float16* pk_efd     = (_Float16*)(ws + 131072);  // 32KB
    _Float16* pk_geo1    = (_Float16*)(ws + 163840);  // 16KB
    _Float16* pk_geo2    = (_Float16*)(ws + 180224);  // 256KB
    _Float16* pk_attn1   = (_Float16*)(ws + 442368);  // 256KB
    _Float16* pk_cls     = (_Float16*)(ws + 704512);  // 16KB

    float* graph  = (float*)d_out;                    // 16*8
    float* nuclei = graph + B_GRAPHS * NUM_CLS;       // N*8
    float* attn   = nuclei + (size_t)N_TOTAL * NUM_CLS; // N

    pack_kernel<<<128, 256, 0, stream>>>(efd_w,   pk_efd,   32,  512, 32);
    pack_kernel<<<64,  256, 0, stream>>>(geo_w1,  pk_geo1,  32,  256, 16);
    pack_kernel<<<512, 256, 0, stream>>>(geo_w2,  pk_geo2,  256, 512, 32);
    pack_kernel<<<512, 256, 0, stream>>>(attn_w1, pk_attn1, 512, 256, 16);
    pack_kernel<<<64,  256, 0, stream>>>(cls_w,   pk_cls,   512, 8,   1);

    bn_partial<<<BN_BLOCKS, 256, 0, stream>>>(x, bnpart);
    bn_final<<<1, 32, 0, stream>>>(bnpart, bn_gamma, bn_beta, scaleshift);

    fused_kernel<<<N_TOTAL / 16 / WAVES, TPB, 0, stream>>>(
        x, scaleshift, pk_efd, pk_geo1, pk_geo2, pk_attn1, pk_cls,
        efd_b, geo_b1, geo_b2, fnw, cls_b, attn_b1, attn_w2, attn_b2,
        nuclei, attn);

    seg_kernel<<<B_GRAPHS, K3T, 0, stream>>>(seq_lens, nuclei, attn, graph);
}
